// DeepSeekV3Router_3659312136540
// MI455X (gfx1250) — compile-verified, matrix operand order checked by reading, not run on hardware
//
#include <hip/hip_runtime.h>

#define T_TOK 8192
#define DDIM  7168
#define EEXP  256
#define KCHUNK 128                // K staged per barrier
#define NKC   (DDIM / KCHUNK)     // 56 chunks
#define NTILES (EEXP / 16)        // 16 n-tiles
#define KSTEP_ELEMS (NTILES * 32 * 16)     // packed-B elems per K=32 step (8192)
#define CHUNK_B_ELEMS (4 * KSTEP_ELEMS)    // packed-B elems per chunk (32768)

typedef __attribute__((ext_vector_type(16))) __bf16 v16bf;
typedef __attribute__((ext_vector_type(8)))  float  v8f;

__device__ __forceinline__ unsigned short f2bf(float f) {
    __bf16 h = (__bf16)f;
    return __builtin_bit_cast(unsigned short, h);
}

__device__ __forceinline__ float sigmoidf(float z) {
    return 1.0f / (1.0f + __expf(-z));
}

// one 16B-per-lane async global->LDS copy (tracked by ASYNCcnt, not LOADcnt)
__device__ __forceinline__ void async_b128(unsigned ldsoff, const void* gp) {
    asm volatile("global_load_async_to_lds_b128 %0, %1, off"
                 :: "v"(ldsoff), "v"(gp) : "memory");
}

// build a 16-bf16 A fragment from 16 contiguous fp32 in LDS (v_cvt_pk_bf16_f32)
__device__ __forceinline__ v16bf cvt_frag(const float* af) {
    v16bf a;
#pragma unroll
    for (int q = 0; q < 4; ++q) {
        float4 f = *(const float4*)(af + q * 4);
        a[q * 4 + 0] = (__bf16)f.x;
        a[q * 4 + 1] = (__bf16)f.y;
        a[q * 4 + 2] = (__bf16)f.z;
        a[q * 4 + 3] = (__bf16)f.w;
    }
    return a;
}

// ---------------------------------------------------------------------------
// Kernel 1: pack W (fp32, [D][E] row-major) into bf16 WMMA-B-operand order.
// tile = kt*16 + ntile; lane L of a tile holds 16 contiguous bf16:
//   n = ntile*16 + (L&15), k = kt*32 + (L>>4)*16 + j  (j = 0..15)
// A K=128 chunk (4 consecutive kt) is one contiguous 64 KB region.
// ---------------------------------------------------------------------------
__global__ __launch_bounds__(256) void pack_w_kernel(
    const float* __restrict__ W, unsigned short* __restrict__ Wp) {
    int o = blockIdx.x * 256 + threadIdx.x;
    if (o >= DDIM * EEXP) return;
    int j     = o & 15;
    int L     = (o >> 4) & 31;
    int tile  = o >> 9;
    int ntile = tile & 15;
    int kt    = tile >> 4;
    int n = ntile * 16 + (L & 15);
    int k = kt * 32 + (L >> 4) * 16 + j;
    Wp[o] = f2bf(W[(size_t)k * EEXP + n]);
}

// ---------------------------------------------------------------------------
// Kernel 2: scores = sigmoid(x @ W).
// Block: 256 thr = 8 waves; tile M=32 x N=256; wave (mw,nw) owns 16 rows x
// 64 cols (4 f32 16x16 accumulators). Both x (fp32) and the packed bf16 B
// panel are staged into double-buffered LDS one chunk ahead via
// global_load_async_to_lds_b128 (ASYNCcnt); the compute loop touches only
// LDS (ds_load + cvt + wmma). One s_wait_asynccnt + barrier per K=128.
// ---------------------------------------------------------------------------
__global__ __launch_bounds__(256) void gemm_sigmoid_kernel(
    const float* __restrict__ x, const unsigned short* __restrict__ Wp,
    float* __restrict__ scores) {
    __shared__ __align__(16) float          ldsA[2][32 * 128];     // 2 x 16 KB
    __shared__ __align__(16) unsigned short ldsB[2][CHUNK_B_ELEMS];// 2 x 64 KB

    const int tid  = threadIdx.x;
    const int lane = tid & 31;
    const int wid  = tid >> 5;
    const int mw   = wid & 1;
    const int nw   = wid >> 1;
    const int mBase = blockIdx.x * 32;

    v8f acc0 = {}, acc1 = {}, acc2 = {}, acc3 = {};

    // A staging: thread -> (row r, 16-float chunk cg), 64B per thread/chunk
    const int r  = tid >> 3;   // 0..31
    const int cg = tid & 7;    // 0..7
    const float* xp = x + (size_t)(mBase + r) * DDIM + cg * 16;
    // generic shared pointers: low 32 bits are the LDS byte offset
    const unsigned ldsA0 = (unsigned)(size_t)&ldsA[0][r * 128 + cg * 16];
    const unsigned ldsA1 = (unsigned)(size_t)&ldsA[1][r * 128 + cg * 16];
    // B staging: linear 64 KB memcpy; thread t copies 16B at q*4096 + t*16
    const unsigned ldsB0 = (unsigned)(size_t)&ldsB[0][0] + tid * 16;
    const unsigned ldsB1 = (unsigned)(size_t)&ldsB[1][0] + tid * 16;

    // stage chunk c into buffer (c&1)
    auto stage_chunk = [&](int c) {
        const unsigned aoff = (c & 1) ? ldsA1 : ldsA0;
        const float*   ap   = xp + (size_t)c * KCHUNK;
#pragma unroll
        for (int q = 0; q < 4; ++q)
            async_b128(aoff + q * 16, ap + q * 4);
        const unsigned boff = (c & 1) ? ldsB1 : ldsB0;
        const char*    bp   = (const char*)(Wp + (size_t)c * CHUNK_B_ELEMS) + tid * 16;
#pragma unroll
        for (int q = 0; q < 16; ++q)
            async_b128(boff + q * 4096, bp + q * 4096);
    };

    // prologue: stage chunk 0
    stage_chunk(0);
    asm volatile("s_wait_asynccnt 0x0" ::: "memory");
    __syncthreads();

    for (int c = 0; c < NKC; ++c) {
        const int cur = c & 1;

        // kick off async staging of chunk c+1 into the other buffer
        if (c + 1 < NKC) stage_chunk(c + 1);

        // compute 4 K-steps (K=32 each) entirely out of LDS
#pragma unroll
        for (int kk = 0; kk < 4; ++kk) {
            // A fragment: row = mw*16 + (lane&15), K-half = lane>>4
            const float* af = &ldsA[cur][(mw * 16 + (lane & 15)) * 128 +
                                         kk * 32 + (lane >> 4) * 16];
            v16bf a = cvt_frag(af);

            // B fragments: tile = kk*16 + nw*4 + j, 512 elems (1 KB) per tile
            const unsigned short* bt =
                &ldsB[cur][(kk * 16 + nw * 4) * 512 + lane * 16];
            v16bf b0 = *(const v16bf*)(bt);
            v16bf b1 = *(const v16bf*)(bt + 1 * 512);
            v16bf b2 = *(const v16bf*)(bt + 2 * 512);
            v16bf b3 = *(const v16bf*)(bt + 3 * 512);

            acc0 = __builtin_amdgcn_wmma_f32_16x16x32_bf16(false, a, false, b0,
                                                           (short)0, acc0, false, false);
            acc1 = __builtin_amdgcn_wmma_f32_16x16x32_bf16(false, a, false, b1,
                                                           (short)0, acc1, false, false);
            acc2 = __builtin_amdgcn_wmma_f32_16x16x32_bf16(false, a, false, b2,
                                                           (short)0, acc2, false, false);
            acc3 = __builtin_amdgcn_wmma_f32_16x16x32_bf16(false, a, false, b3,
                                                           (short)0, acc3, false, false);
        }

        // chunk c+1 LDS writes must land before anyone reads them
        asm volatile("s_wait_asynccnt 0x0" ::: "memory");
        __syncthreads();
    }

    // Epilogue: C/D layout -> VGPR rr: lanes 0-15 M=rr, lanes 16-31 M=rr+8;
    // N = lane & 15 within each 16-wide tile.
    const int mRow = mBase + mw * 16 + (lane >> 4) * 8;
    const int nCol = nw * 64 + (lane & 15);
#pragma unroll
    for (int rr = 0; rr < 8; ++rr) {
        size_t base = (size_t)(mRow + rr) * EEXP + nCol;
        scores[base + 0]  = sigmoidf(acc0[rr]);
        scores[base + 16] = sigmoidf(acc1[rr]);
        scores[base + 32] = sigmoidf(acc2[rr]);
        scores[base + 48] = sigmoidf(acc3[rr]);
    }
}

// ---------------------------------------------------------------------------
// Kernel 3: grouped top-k routing. One wave32 per token; lane l owns experts
// l*8 .. l*8+7 (group g = l>>2, 32 experts per group, 8 groups).
// ---------------------------------------------------------------------------
__global__ __launch_bounds__(256) void route_kernel(
    const float* __restrict__ scores, const float* __restrict__ bias,
    float* __restrict__ outW, int* __restrict__ outI) {
    const int lane = threadIdx.x & 31;
    const int wid  = threadIdx.x >> 5;
    const int t    = blockIdx.x * 8 + wid;
    const float* row = scores + (size_t)t * EEXP;

    float p[8], s[8];
#pragma unroll
    for (int j = 0; j < 8; ++j) {
        p[j] = row[lane * 8 + j];
        s[j] = p[j] + bias[lane * 8 + j];
    }

    // per-lane top-2 of biased scores
    float a1 = -1e30f, a2 = -1e30f;
#pragma unroll
    for (int j = 0; j < 8; ++j) {
        float v = s[j];
        if (v > a1) { a2 = a1; a1 = v; } else if (v > a2) { a2 = v; }
    }
    // merge across the 4 lanes of the group
#pragma unroll
    for (int off = 1; off <= 2; off <<= 1) {
        float b1 = __shfl_xor(a1, off, 32);
        float b2 = __shfl_xor(a2, off, 32);
        if (b1 > a1) { a2 = fmaxf(a1, b2); a1 = b1; }
        else         { a2 = fmaxf(a2, b1); }
    }
    float gscore = a1 + a2;               // same on all 4 lanes of group
    const int g = lane >> 2;

    // rank my group among the 8 group scores (lower index wins ties)
    float gs[8];
#pragma unroll
    for (int gg = 0; gg < 8; ++gg) gs[gg] = __shfl(gscore, gg * 4, 32);
    int rank = 0;
#pragma unroll
    for (int gg = 0; gg < 8; ++gg)
        if (gs[gg] > gs[g] || (gs[gg] == gs[g] && gg < g)) rank++;
    const bool keep = (rank < 4);         // TOPK_GROUPS = 4

    float v[8];
#pragma unroll
    for (int j = 0; j < 8; ++j) v[j] = keep ? s[j] : 0.0f;  // ref: masked -> 0.0

    float wout[8]; int iout[8]; float wsum = 0.0f;
#pragma unroll
    for (int k = 0; k < 8; ++k) {
        float bv = -1e30f; int bi = 0;
#pragma unroll
        for (int j = 0; j < 8; ++j) {
            float c = v[j];
            int   ci = lane * 8 + j;
            if (c > bv || (c == bv && ci < bi)) { bv = c; bi = ci; }
        }
#pragma unroll
        for (int off = 16; off >= 1; off >>= 1) {
            float ov = __shfl_xor(bv, off, 32);
            int   oi = __shfl_xor(bi, off, 32);
            if (ov > bv || (ov == bv && oi < bi)) { bv = ov; bi = oi; }
        }
        // fetch unbiased weight p at winning expert (bi uniform across wave)
        float myp = 0.0f;
#pragma unroll
        for (int j = 0; j < 8; ++j) if ((bi & 7) == j) myp = p[j];
        float wv = __shfl(myp, bi >> 3, 32);
        wout[k] = wv; iout[k] = bi; wsum += wv;
        // remove winner
#pragma unroll
        for (int j = 0; j < 8; ++j)
            if ((bi >> 3) == lane && (bi & 7) == j) v[j] = -1e30f;
    }

    const float scale = 2.5f / (wsum + 1e-20f);   // norm_topk + routed scaling
    if (lane < 8) {
        float wsel = 0.0f; int isel = 0;
#pragma unroll
        for (int k = 0; k < 8; ++k)
            if (k == lane) { wsel = wout[k]; isel = iout[k]; }
        outW[(size_t)t * 8 + lane] = wsel * scale;
        outI[(size_t)t * 8 + lane] = isel;
    }
}

// ---------------------------------------------------------------------------
extern "C" void kernel_launch(void* const* d_in, const int* in_sizes, int n_in,
                              void* d_out, int out_size, void* d_ws, size_t ws_size,
                              hipStream_t stream) {
    const float* x    = (const float*)d_in[0];   // [T, D] fp32
    const float* W    = (const float*)d_in[1];   // [D, E] fp32
    const float* bias = (const float*)d_in[2];   // [E]    fp32

    float*          scores = (float*)d_ws;                                   // 8 MB
    unsigned short* Wp     = (unsigned short*)((char*)d_ws +
                              (size_t)T_TOK * EEXP * sizeof(float));         // 3.7 MB

    float* outW = (float*)d_out;                       // [T, 8] fp32 weights
    int*   outI = (int*)d_out + (size_t)T_TOK * 8;     // [T, 8] int32 indices

    pack_w_kernel<<<(DDIM * EEXP + 255) / 256, 256, 0, stream>>>(W, Wp);
    gemm_sigmoid_kernel<<<T_TOK / 32, 256, 0, stream>>>(x, Wp, scores);
    route_kernel<<<T_TOK / 8, 256, 0, stream>>>(scores, bias, outW, outI);
}
